// GroupAttention_44547400794562
// MI455X (gfx1250) — compile-verified, matrix operand order checked by reading, not run on hardware
//
#include <hip/hip_runtime.h>
#include <hip/hip_bf16.h>
#include <stdint.h>
#include <math.h>

#define B_ 8
#define S_ 1024
#define D_ 1024
#define MBLK 128
#define NBLK 128
#define KSTEP 32

typedef __attribute__((ext_vector_type(16))) __bf16       v16bf;
typedef __attribute__((ext_vector_type(8)))  float        v8f;
typedef __attribute__((ext_vector_type(4)))  unsigned int u32x4;
typedef __attribute__((ext_vector_type(4)))  int          v4i;
typedef __attribute__((address_space(1))) v4i as1_v4i;    // global
typedef __attribute__((address_space(3))) v4i as3_v4i;    // LDS

union Frag16 { v16bf v; u32x4 u[2]; };

// ---------------------------------------------------------------------------
// Async global->LDS (CDNA5 ASYNCcnt path) with sync fallback
// ---------------------------------------------------------------------------
__device__ __forceinline__ void async_b128(const __bf16* g, __bf16* l) {
#if __has_builtin(__builtin_amdgcn_global_load_async_to_lds_b128)
  __builtin_amdgcn_global_load_async_to_lds_b128(
      (as1_v4i*)(void*)g, (as3_v4i*)l, 0, 0);
#else
  *(u32x4*)l = *(const u32x4*)g;
#endif
}

__device__ __forceinline__ void wait_async() {
#if __has_builtin(__builtin_amdgcn_s_wait_asynccnt)
  __builtin_amdgcn_s_wait_asynccnt(0);
#elif __has_builtin(__builtin_amdgcn_global_load_async_to_lds_b128)
  asm volatile("s_wait_asynccnt 0x0" ::: "memory");
#endif
}

// ---------------------------------------------------------------------------
// Stage one 128x32 A-tile and 128x32 B-tile into LDS (16 KB per K-step).
// 256 threads x 2 x 16B chunks per tile.
// ---------------------------------------------------------------------------
__device__ __forceinline__ void stage_tile(const __bf16* __restrict__ Ag,
                                           const __bf16* __restrict__ Bg,
                                           __bf16* As, __bf16* Bs, int kk) {
#pragma unroll
  for (int it = 0; it < 2; ++it) {
    const int e = threadIdx.x + it * 256;          // 0..511
    const int row = e >> 2, seg = e & 3;           // row 0..127, 16B segment
    async_b128(Ag + (size_t)row * D_ + kk + seg * 8, As + row * KSTEP + seg * 8);
  }
#pragma unroll
  for (int it = 0; it < 2; ++it) {
    const int e = threadIdx.x + it * 256;
    const int row = e >> 2, seg = e & 3;
    async_b128(Bg + (size_t)row * D_ + kk + seg * 8, Bs + row * KSTEP + seg * 8);
  }
}

// ---------------------------------------------------------------------------
// Per-wave 64x32 sub-tile: 4x2 WMMA accumulators fed from LDS (ds_load_b128)
// ---------------------------------------------------------------------------
__device__ __forceinline__ void mma_tile(const __bf16* As, const __bf16* Bs,
                                         int wm, int wn, int lm, int hi,
                                         v8f acc[4][2]) {
  Frag16 a[4], b[2];
#pragma unroll
  for (int fm = 0; fm < 4; ++fm) {
    const __bf16* p = As + (wm * 64 + fm * 16 + lm) * KSTEP + hi * 8;
    a[fm].u[0] = *(const u32x4*)p;
    a[fm].u[1] = *(const u32x4*)(p + 16);
  }
#pragma unroll
  for (int fn = 0; fn < 2; ++fn) {
    const __bf16* p = Bs + (wn * 32 + fn * 16 + lm) * KSTEP + hi * 8;
    b[fn].u[0] = *(const u32x4*)p;
    b[fn].u[1] = *(const u32x4*)(p + 16);
  }
#pragma unroll
  for (int fm = 0; fm < 4; ++fm)
#pragma unroll
    for (int fn = 0; fn < 2; ++fn)
      acc[fm][fn] = __builtin_amdgcn_wmma_f32_16x16x32_bf16(
          false, a[fm].v, false, b[fn].v, (short)0, acc[fm][fn], false, false);
}

// ---------------------------------------------------------------------------
// 1) LayerNorm (unbiased std, eps on std) -> bf16 activations
// ---------------------------------------------------------------------------
__global__ void ln_bf16_kernel(const float* __restrict__ ctx,
                               const float* __restrict__ gamma,
                               const float* __restrict__ beta,
                               __bf16* __restrict__ xbf) {
  const int row = blockIdx.x;                 // b*S + s
  const int tid = threadIdx.x;                // 256 threads
  const float* p = ctx + (size_t)row * D_;
  __shared__ float s1[256];
  __shared__ float s2[256];
  float v[4]; float sum = 0.f, sq = 0.f;
#pragma unroll
  for (int t = 0; t < 4; ++t) {
    v[t] = p[tid + t * 256];
    sum += v[t]; sq += v[t] * v[t];
  }
  s1[tid] = sum; s2[tid] = sq; __syncthreads();
  for (int off = 128; off > 0; off >>= 1) {
    if (tid < off) { s1[tid] += s1[tid + off]; s2[tid] += s2[tid + off]; }
    __syncthreads();
  }
  const float mu  = s1[0] * (1.f / D_);
  const float var = (s2[0] - (float)D_ * mu * mu) * (1.f / (float)(D_ - 1));
  const float inv = 1.f / (sqrtf(var) + 1e-6f);
#pragma unroll
  for (int t = 0; t < 4; ++t) {
    const int c = tid + t * 256;
    const float xn = gamma[c] * (v[t] - mu) * inv + beta[c];
    xbf[(size_t)row * D_ + c] = (__bf16)xn;
  }
}

// ---------------------------------------------------------------------------
// 2) Weights f32 -> bf16
// ---------------------------------------------------------------------------
__global__ void w_to_bf16_kernel(const float* __restrict__ wk,
                                 const float* __restrict__ wq,
                                 __bf16* __restrict__ wkb,
                                 __bf16* __restrict__ wqb) {
  const int i = blockIdx.x * blockDim.x + threadIdx.x;
  if (i < D_ * D_) { wkb[i] = (__bf16)wk[i]; wqb[i] = (__bf16)wq[i]; }
}

// ---------------------------------------------------------------------------
// 3) Projections: key = x@Wk.T + bk ; query = x@Wq.T + bq  (grid.z selects)
//    128x128 block tile, LDS double-buffered, async staging.
// ---------------------------------------------------------------------------
__global__ void __launch_bounds__(256)
proj_gemm_kernel(const __bf16* __restrict__ X,
                 const __bf16* __restrict__ Wkb,
                 const __bf16* __restrict__ Wqb,
                 const float* __restrict__ bk,
                 const float* __restrict__ bq,
                 __bf16* __restrict__ keyb,
                 __bf16* __restrict__ qryb) {
  __shared__ __align__(16) __bf16 As[2][MBLK * KSTEP];
  __shared__ __align__(16) __bf16 Bs[2][NBLK * KSTEP];
  const int wave = threadIdx.x >> 5, lane = threadIdx.x & 31;
  const int lm = lane & 15, hi = lane >> 4;
  const int wm = wave >> 2, wn = wave & 3;
  const int m0 = blockIdx.y * MBLK;
  const int n0 = blockIdx.x * NBLK;
  const __bf16* W    = blockIdx.z ? Wqb : Wkb;
  const float*  bias = blockIdx.z ? bq  : bk;
  __bf16*       Out  = blockIdx.z ? qryb : keyb;
  const __bf16* Ag = X + (size_t)m0 * D_;
  const __bf16* Bg = W + (size_t)n0 * D_;

  v8f acc[4][2] = {};
  stage_tile(Ag, Bg, As[0], Bs[0], 0);
  for (int kt = 0; kt < D_ / KSTEP; ++kt) {
    const int cur = kt & 1;
    wait_async();
    __syncthreads();
    if (kt + 1 < D_ / KSTEP)
      stage_tile(Ag, Bg, As[cur ^ 1], Bs[cur ^ 1], (kt + 1) * KSTEP);
    mma_tile(As[cur], Bs[cur], wm, wn, lm, hi, acc);
  }
#pragma unroll
  for (int fn = 0; fn < 2; ++fn) {
    const int n = n0 + wn * 32 + fn * 16 + lm;
    const float bcol = bias[n];
#pragma unroll
    for (int fm = 0; fm < 4; ++fm) {
      const int mbase = m0 + wm * 64 + fm * 16 + hi * 8;
#pragma unroll
      for (int r = 0; r < 8; ++r)
        Out[(size_t)(mbase + r) * D_ + n] = (__bf16)(acc[fm][fn][r] + bcol);
    }
  }
}

// ---------------------------------------------------------------------------
// 4) scores[b,q,k] = dot(query[b,q,:], key[b,k,:]) / 512   (same tiling)
// ---------------------------------------------------------------------------
__global__ void __launch_bounds__(256)
scores_gemm_kernel(const __bf16* __restrict__ qryb,
                   const __bf16* __restrict__ keyb,
                   float* __restrict__ scores) {
  __shared__ __align__(16) __bf16 As[2][MBLK * KSTEP];
  __shared__ __align__(16) __bf16 Bs[2][NBLK * KSTEP];
  const int wave = threadIdx.x >> 5, lane = threadIdx.x & 31;
  const int lm = lane & 15, hi = lane >> 4;
  const int wm = wave >> 2, wn = wave & 3;
  const int b  = blockIdx.z;
  const int q0 = blockIdx.y * MBLK;
  const int k0 = blockIdx.x * NBLK;
  const __bf16* Ag = qryb + ((size_t)b * S_ + q0) * D_;
  const __bf16* Bg = keyb + ((size_t)b * S_ + k0) * D_;

  v8f acc[4][2] = {};
  stage_tile(Ag, Bg, As[0], Bs[0], 0);
  for (int kt = 0; kt < D_ / KSTEP; ++kt) {
    const int cur = kt & 1;
    wait_async();
    __syncthreads();
    if (kt + 1 < D_ / KSTEP)
      stage_tile(Ag, Bg, As[cur ^ 1], Bs[cur ^ 1], (kt + 1) * KSTEP);
    mma_tile(As[cur], Bs[cur], wm, wn, lm, hi, acc);
  }
  float* out = scores + (size_t)b * S_ * S_;
#pragma unroll
  for (int fn = 0; fn < 2; ++fn) {
    const int k = k0 + wn * 32 + fn * 16 + lm;
#pragma unroll
    for (int fm = 0; fm < 4; ++fm) {
      const int qbase = q0 + wm * 64 + fm * 16 + hi * 8;
#pragma unroll
      for (int r = 0; r < 8; ++r)
        out[(size_t)(qbase + r) * S_ + k] = acc[fm][fn][r] * (1.f / 512.f);
    }
  }
}

// ---------------------------------------------------------------------------
// 5) adj-masked row softmax (in place on scores)
// ---------------------------------------------------------------------------
__global__ void masked_softmax_kernel(float* __restrict__ scores,
                                      const int* __restrict__ adj) {
  const int row = blockIdx.x;                  // b*S + q
  const int tid = threadIdx.x;
  float* s = scores + (size_t)row * S_;
  const int* a = adj + (size_t)row * S_;
  __shared__ float red[256];
  float v[4]; float mx = -3.4e38f;
#pragma unroll
  for (int t = 0; t < 4; ++t) {
    const int c = tid + t * 256;
    v[t] = (a[c] == 0) ? -1e9f : s[c];
    mx = fmaxf(mx, v[t]);
  }
  red[tid] = mx; __syncthreads();
  for (int off = 128; off > 0; off >>= 1) {
    if (tid < off) red[tid] = fmaxf(red[tid], red[tid + off]);
    __syncthreads();
  }
  mx = red[0]; __syncthreads();
  float sum = 0.f;
#pragma unroll
  for (int t = 0; t < 4; ++t) { v[t] = expf(v[t] - mx); sum += v[t]; }
  red[tid] = sum; __syncthreads();
  for (int off = 128; off > 0; off >>= 1) {
    if (tid < off) red[tid] += red[tid + off];
    __syncthreads();
  }
  const float inv = 1.f / red[0];
#pragma unroll
  for (int t = 0; t < 4; ++t) s[tid + t * 256] = v[t] * inv;
}

// ---------------------------------------------------------------------------
// 6) neibor = prior + (1-prior)*sqrt(attn * attn^T + 1e-9)  -> d_out[half 2]
// ---------------------------------------------------------------------------
__global__ void neibor_kernel(const float* __restrict__ attn,
                              const float* __restrict__ prior_p,
                              float* __restrict__ neib) {
  const int row = blockIdx.x;                  // b*S + i
  const int b = row >> 10, i = row & (S_ - 1);
  const float prior = prior_p[0];
  const float* ab = attn + (size_t)b * S_ * S_;
  float* out = neib + (size_t)row * S_;
#pragma unroll
  for (int t = 0; t < 4; ++t) {
    const int j = threadIdx.x + t * 256;
    const float x = ab[(size_t)i * S_ + j] * ab[(size_t)j * S_ + i];
    out[j] = prior + (1.f - prior) * sqrtf(x + 1e-9f);
  }
}

// ---------------------------------------------------------------------------
// 7) cum[b,j] = sum_{k<j} log(neibor[b,k,k+1] + 1e-9)   (Hillis-Steele scan)
//    (Replaces both tri-matmuls from the reference.)
// ---------------------------------------------------------------------------
__global__ void prefix_kernel(const float* __restrict__ neib,
                              float* __restrict__ cum) {
  const int b = blockIdx.x;
  const int j = threadIdx.x;                   // 1024 threads
  __shared__ float buf[S_];
  float e = 0.f;
  if (j > 0) {
    const float a = neib[((size_t)b * S_ + (j - 1)) * S_ + j];  // superdiag [j-1, j]
    e = logf(a + 1e-9f);
  }
  buf[j] = e; __syncthreads();
  for (int off = 1; off < S_; off <<= 1) {
    const float add = (j >= off) ? buf[j - off] : 0.f;
    __syncthreads();
    buf[j] += add;
    __syncthreads();
  }
  cum[b * S_ + j] = buf[j];
}

// ---------------------------------------------------------------------------
// 8) g_attn[b,i,j] = (i==j) ? neibor[b,i,i]
//                           : exp(cum[max(i,j)] - cum[min(i,j)]) + 1e-9
// ---------------------------------------------------------------------------
__global__ void gattn_kernel(const float* __restrict__ neib,
                             const float* __restrict__ cum,
                             float* __restrict__ g) {
  const int row = blockIdx.x;                  // b*S + i
  const int b = row >> 10, i = row & (S_ - 1);
  const float ci = cum[b * S_ + i];
  const float diag = neib[((size_t)b * S_ + i) * S_ + i];
  float* out = g + (size_t)row * S_;
#pragma unroll
  for (int t = 0; t < 4; ++t) {
    const int j = threadIdx.x + t * 256;
    float val;
    if (j == i) {
      val = diag;
    } else {
      const float cj  = cum[b * S_ + j];
      const float chi = (j > i) ? cj : ci;     // cum at larger index
      const float clo = (j > i) ? ci : cj;     // cum at smaller index
      val = expf(chi - clo) + 1e-9f;
    }
    out[j] = val;
  }
}

// ---------------------------------------------------------------------------
extern "C" void kernel_launch(void* const* d_in, const int* in_sizes, int n_in,
                              void* d_out, int out_size, void* d_ws, size_t ws_size,
                              hipStream_t stream) {
  (void)in_sizes; (void)n_in; (void)out_size; (void)ws_size;
  const float* ctx   = (const float*)d_in[0];
  /* d_in[1] eos_mask unused by the reference */
  const float* prior = (const float*)d_in[2];
  const int*   adj   = (const int*)d_in[3];
  const float* wk    = (const float*)d_in[4];
  const float* bk    = (const float*)d_in[5];
  const float* wq    = (const float*)d_in[6];
  const float* bq    = (const float*)d_in[7];
  const float* gamma = (const float*)d_in[8];
  const float* beta  = (const float*)d_in[9];

  float* g_out = (float*)d_out;                        // (B,S,S) g_attn
  float* neib  = g_out + (size_t)B_ * S_ * S_;         // (B,S,S) neibor_attn

  char* ws = (char*)d_ws;
  size_t off = 0;
  auto alloc = [&](size_t bytes) -> void* {
    void* p = ws + off;
    off = (off + bytes + 255) & ~(size_t)255;
    return p;
  };
  __bf16* xbf  = (__bf16*)alloc((size_t)B_ * S_ * D_ * 2);
  __bf16* wkb  = (__bf16*)alloc((size_t)D_ * D_ * 2);
  __bf16* wqb  = (__bf16*)alloc((size_t)D_ * D_ * 2);
  __bf16* keyb = (__bf16*)alloc((size_t)B_ * S_ * D_ * 2);
  __bf16* qryb = (__bf16*)alloc((size_t)B_ * S_ * D_ * 2);
  float*  sc   = (float*)alloc((size_t)B_ * S_ * S_ * 4);
  float*  cum  = (float*)alloc((size_t)B_ * S_ * 4);

  ln_bf16_kernel<<<B_ * S_, 256, 0, stream>>>(ctx, gamma, beta, xbf);
  w_to_bf16_kernel<<<(D_ * D_ + 255) / 256, 256, 0, stream>>>(wk, wq, wkb, wqb);

  // proj: M = B*S = 8192, N = 1024, 128x128 block tiles
  proj_gemm_kernel<<<dim3(D_ / NBLK, (B_ * S_) / MBLK, 2), 256, 0, stream>>>(
      xbf, wkb, wqb, bk, bq, keyb, qryb);

  // scores: per batch 1024x1024, 128x128 block tiles
  scores_gemm_kernel<<<dim3(S_ / NBLK, S_ / MBLK, B_), 256, 0, stream>>>(
      qryb, keyb, sc);

  masked_softmax_kernel<<<B_ * S_, 256, 0, stream>>>(sc, adj);
  neibor_kernel<<<B_ * S_, 256, 0, stream>>>(sc, prior, neib);
  prefix_kernel<<<B_, S_, 0, stream>>>(neib, cum);
  gattn_kernel<<<B_ * S_, 256, 0, stream>>>(neib, cum, g_out);
}
